// SwinTransformerBlock_10402410791774
// MI455X (gfx1250) — matrix operand order repaired and checked by reading, not compiled, not run on hardware
//
#include <hip/hip_runtime.h>

// ---------------------------------------------------------------------------
// Swin block (B=1, H=W=64, C=256, heads=8, ws=8, shift=4, hid=1024) for
// gfx1250 / MI455X.  All matmuls run on v_wmma_f32_16x16x32_bf16 (wave32).
// Cross-attention (full 4096-token softmax attention) is implemented as
// flash attention so the 536MB score matrix never touches HBM.
// ---------------------------------------------------------------------------

typedef __attribute__((ext_vector_type(16))) __bf16 v16bf;
typedef __attribute__((ext_vector_type(8)))  __bf16 v8bf;
typedef __attribute__((ext_vector_type(8)))  float  v8f;

#define WMMA_BF16(a, b, c) \
  __builtin_amdgcn_wmma_f32_16x16x32_bf16(false, (a), false, (b), (short)0, (c), false, false)

__device__ __forceinline__ v8f vzero() {
  v8f z;
#pragma unroll
  for (int e = 0; e < 8; ++e) z[e] = 0.0f;
  return z;
}

// Build a 16x32 bf16 A-fragment from two contiguous 8-element runs
// (lane l: M = l&15; K = kb..kb+7 and kb+8..kb+15, kb = (l<16)?0:16).
__device__ __forceinline__ v16bf frag2(const __bf16* p0, const __bf16* p1) {
  v8bf x = *(const v8bf*)p0;
  v8bf y = *(const v8bf*)p1;
  v16bf r;
#pragma unroll
  for (int i = 0; i < 8; ++i) { r[i] = x[i]; r[i + 8] = y[i]; }
  return r;
}

// Window-layout row t -> original (h*64+w) row.  Both the forward
// (shift -4, partition) and inverse (reverse, shift +4) mappings reduce to
// orig = ((hs+4)%64, (ws+4)%64) where (hs,ws) are the window-layout coords.
__device__ __forceinline__ int permrow(int t) {
  int win = t >> 6, pos = t & 63;
  int hs = ((win >> 3) << 3) + (pos >> 3);
  int ws = ((win & 7) << 3) + (pos & 7);
  int h = (hs + 4) & 63, w = (ws + 4) & 63;
  return (h << 6) + w;
}

// ---------------------------------------------------------------------------
__global__ void cvt_bf16(const float* __restrict__ s, __bf16* __restrict__ d, int n) {
  int i = blockIdx.x * blockDim.x + threadIdx.x;
  if (i < n) d[i] = (__bf16)s[i];
}

// ---------------------------------------------------------------------------
// LayerNorm over C=256; one wave per token (8 floats/lane, wave32 shfl tree).
// perm=1: token index t is window-layout, source row is permrow(t) (LN1+shift).
__global__ __launch_bounds__(256) void ln_kernel(
    const float* __restrict__ x, const float* __restrict__ g, const float* __restrict__ b,
    __bf16* __restrict__ outB, float* __restrict__ outF, int perm) {
  int wave = threadIdx.x >> 5, lane = threadIdx.x & 31;
  int t = blockIdx.x * 8 + wave;
  int srow = perm ? permrow(t) : t;
  const float* xr = x + (long)srow * 256;
  float v[8], s = 0.0f, s2 = 0.0f;
#pragma unroll
  for (int j = 0; j < 8; ++j) { v[j] = xr[lane * 8 + j]; s += v[j]; s2 += v[j] * v[j]; }
#pragma unroll
  for (int off = 16; off >= 1; off >>= 1) {
    s += __shfl_xor(s, off, 32);
    s2 += __shfl_xor(s2, off, 32);
  }
  float mu = s * (1.0f / 256.0f);
  float var = s2 * (1.0f / 256.0f) - mu * mu;
  float rs = rsqrtf(var + 1e-5f);
#pragma unroll
  for (int j = 0; j < 8; ++j) {
    int c = lane * 8 + j;
    float y = (v[j] - mu) * rs * g[c] + b[c];
    outB[(long)t * 256 + c] = (__bf16)y;
    if (outF) outF[(long)t * 256 + c] = y;
  }
}

// ---------------------------------------------------------------------------
// Generic bf16 GEMM: C[M,N] = A[M,K] @ B[K,N] (+bias)(+ReLU)(+residual).
// Block tile 128x64, 8 waves in 4x2, wave tile 32x32 (2x2 WMMA), K-step 32.
// B is staged transposed in LDS so B-fragments are contiguous 32B loads.
// perm=1: output row (and residual row) is window-reverse-permuted.
__global__ __launch_bounds__(256) void gemm_bf16(
    const __bf16* __restrict__ A, const __bf16* __restrict__ B, int K, int N,
    const float* __restrict__ bias, const float* __restrict__ resid, int perm,
    float* __restrict__ outF, __bf16* __restrict__ outB, int relu) {
  __shared__ __align__(32) __bf16 Atile[128][32];
  __shared__ __align__(32) __bf16 Bt[64][32];

  const int m0 = blockIdx.y * 128, n0 = blockIdx.x * 64;
  const int wave = threadIdx.x >> 5, lane = threadIdx.x & 31;
  const int wm = wave >> 1, wn = wave & 1;
  const int ml = lane & 15;
  const int kb = (lane < 16) ? 0 : 16;   // A-fragment K base
  const int kb2 = (lane < 16) ? 0 : 16;  // B-fragment K base

  v8f acc[2][2];
#pragma unroll
  for (int i = 0; i < 2; ++i)
#pragma unroll
    for (int j = 0; j < 2; ++j) acc[i][j] = vzero();

  for (int k0 = 0; k0 < K; k0 += 32) {
    __syncthreads();
    // A tile: 128x32 bf16 = 512 16B-chunks, 2 per thread (global_load_b128).
#pragma unroll
    for (int c = 0; c < 2; ++c) {
      int id = threadIdx.x + c * 256;
      int r = id >> 2, q = id & 3;
      if (k0 + 32 < K)
        __builtin_prefetch(&A[(long)(m0 + r) * K + k0 + 32 + q * 8], 0, 1);
      *(v8bf*)&Atile[r][q * 8] = *(const v8bf*)&A[(long)(m0 + r) * K + k0 + q * 8];
    }
    // B tile transposed: row-major [k][n] -> Bt[n][k].
    {
      int kr = threadIdx.x >> 3, q = threadIdx.x & 7;
      v8bf v = *(const v8bf*)&B[(long)(k0 + kr) * N + n0 + q * 8];
#pragma unroll
      for (int j = 0; j < 8; ++j) Bt[q * 8 + j][kr] = v[j];
    }
    __syncthreads();

    v16bf a[2], b[2];
    a[0] = frag2(&Atile[wm * 32 + ml][kb], &Atile[wm * 32 + ml][kb + 8]);
    a[1] = frag2(&Atile[wm * 32 + 16 + ml][kb], &Atile[wm * 32 + 16 + ml][kb + 8]);
    b[0] = *(const v16bf*)&Bt[wn * 32 + ml][kb2];
    b[1] = *(const v16bf*)&Bt[wn * 32 + 16 + ml][kb2];
#pragma unroll
    for (int i = 0; i < 2; ++i)
#pragma unroll
      for (int j = 0; j < 2; ++j) acc[i][j] = WMMA_BF16(a[i], b[j], acc[i][j]);
  }

  // Epilogue.  C-layout: lane l owns N = n0+..+(l&15); VGPR r owns row r or 8+r.
  const int rb = (lane < 16) ? 0 : 8;
#pragma unroll
  for (int i = 0; i < 2; ++i) {
#pragma unroll
    for (int j = 0; j < 2; ++j) {
      int ng = n0 + wn * 32 + j * 16 + ml;
      float bv = bias ? bias[ng] : 0.0f;
#pragma unroll
      for (int r = 0; r < 8; ++r) {
        int mg = m0 + wm * 32 + i * 16 + rb + r;
        float v = acc[i][j][r] + bv;
        if (relu) v = fmaxf(v, 0.0f);
        int orow = perm ? permrow(mg) : mg;
        long idx = (long)orow * N + ng;
        if (resid) v += resid[idx];
        if (outF) outF[idx] = v;
        if (outB) outB[idx] = (__bf16)v;
      }
    }
  }
}

// ---------------------------------------------------------------------------
// Flash attention on packed qkv [rows][3*256] bf16 (q|k|v, head-major d=32).
// grid = (seq/64 m-blocks, 8 heads, n_sequences); 4 waves x 16 rows per block.
// Online softmax state lives per-lane for the 8 rows each half-wave owns in
// the WMMA C-layout; reductions are 16-lane shfl_xor trees.
__global__ __launch_bounds__(128) void flash_attn(
    const __bf16* __restrict__ qkv, __bf16* __restrict__ out,
    const float* __restrict__ scale_p, int seqLen) {
  __shared__ __align__(32) __bf16 Klds[64][32];     // [token][ch]
  __shared__ __align__(32) __bf16 Vt[32][64];       // [ch][token]
  __shared__ __align__(32) __bf16 Plds[4][16][64];  // per-wave P tile

  const int head = blockIdx.y;
  const long base = (long)blockIdx.z * seqLen;
  const int wave = threadIdx.x >> 5, lane = threadIdx.x & 31;
  const int row0 = blockIdx.x * 64 + wave * 16;
  const float scale = scale_p[0];
  const int ml = lane & 15;
  const int kb = (lane < 16) ? 0 : 16;
  const int kb2 = (lane < 16) ? 0 : 16;
  const int rb = (lane < 16) ? 0 : 8;

  // Q fragment for this wave's 16 rows (K = d = 32 -> single A fragment).
  long qoff = (base + row0 + ml) * 768 + head * 32;
  v16bf aq = frag2(&qkv[qoff + kb], &qkv[qoff + kb + 8]);

  float mrun[8], lrun[8];
  v8f o0 = vzero(), o1 = vzero();
#pragma unroll
  for (int r = 0; r < 8; ++r) { mrun[r] = -3.0e38f; lrun[r] = 0.0f; }

  const int nKB = seqLen >> 6;
  for (int kbk = 0; kbk < nKB; ++kbk) {
    __syncthreads();  // previous tile fully consumed by all waves
    for (int id = threadIdx.x; id < 256; id += 128) {
      int tok = id >> 2, q8 = (id & 3) * 8;
      long grow = (base + kbk * 64 + tok) * 768 + 256 + head * 32 + q8;
      *(v8bf*)&Klds[tok][q8] = *(const v8bf*)&qkv[grow];        // K row-major
      v8bf vv = *(const v8bf*)&qkv[grow + 256];                 // V transposed
#pragma unroll
      for (int j = 0; j < 8; ++j) Vt[q8 + j][tok] = vv[j];
    }
    __syncthreads();

    // S = scale * Q K^T : B[k=ch][n=token] = Klds[token][ch] (4 n-tiles).
    v8f s[4];
#pragma unroll
    for (int ni = 0; ni < 4; ++ni) {
      v16bf bk = *(const v16bf*)&Klds[ni * 16 + ml][kb2];
      v8f z = vzero();
      s[ni] = WMMA_BF16(aq, bk, z);
#pragma unroll
      for (int r = 0; r < 8; ++r) s[ni][r] *= scale;
    }

    // Online softmax (rows r / 8+r per half-wave; 16-lane reductions).
    float alpha[8];
#pragma unroll
    for (int r = 0; r < 8; ++r) {
      float mx = fmaxf(fmaxf(s[0][r], s[1][r]), fmaxf(s[2][r], s[3][r]));
#pragma unroll
      for (int off = 8; off >= 1; off >>= 1) mx = fmaxf(mx, __shfl_xor(mx, off, 32));
      float mnew = fmaxf(mrun[r], mx);
      alpha[r] = __expf(mrun[r] - mnew);
      mrun[r] = mnew;
    }
#pragma unroll
    for (int r = 0; r < 8; ++r) {
      float rs = 0.0f;
#pragma unroll
      for (int ni = 0; ni < 4; ++ni) {
        float pv = __expf(s[ni][r] - mrun[r]);
        s[ni][r] = pv;
        rs += pv;
      }
#pragma unroll
      for (int off = 8; off >= 1; off >>= 1) rs += __shfl_xor(rs, off, 32);
      lrun[r] = lrun[r] * alpha[r] + rs;
      o0[r] *= alpha[r];
      o1[r] *= alpha[r];
    }

    // P (C-layout regs) -> per-wave LDS so it can be re-read as A fragments.
#pragma unroll
    for (int ni = 0; ni < 4; ++ni)
#pragma unroll
      for (int r = 0; r < 8; ++r)
        Plds[wave][rb + r][ni * 16 + ml] = (__bf16)s[ni][r];
    __syncthreads();

    // O += P @ V : K = 64 tokens (2 steps), N = d = 32 (2 n-tiles).
#pragma unroll
    for (int ks = 0; ks < 2; ++ks) {
      v16bf ap = frag2(&Plds[wave][ml][ks * 32 + kb], &Plds[wave][ml][ks * 32 + kb + 8]);
      v16bf bv0 = *(const v16bf*)&Vt[ml][ks * 32 + kb2];
      v16bf bv1 = *(const v16bf*)&Vt[16 + ml][ks * 32 + kb2];
      o0 = WMMA_BF16(ap, bv0, o0);
      o1 = WMMA_BF16(ap, bv1, o1);
    }
  }

  // Normalize and store bf16 (feeds the projection GEMM).
#pragma unroll
  for (int r = 0; r < 8; ++r) {
    float inv = 1.0f / lrun[r];
    long t = base + row0 + rb + r;
    out[t * 256 + head * 32 + ml] = (__bf16)(o0[r] * inv);
    out[t * 256 + head * 32 + 16 + ml] = (__bf16)(o1[r] * inv);
  }
}

// ---------------------------------------------------------------------------
extern "C" void kernel_launch(void* const* d_in, const int* in_sizes, int n_in,
                              void* d_out, int out_size, void* d_ws, size_t ws_size,
                              hipStream_t stream) {
  (void)in_sizes; (void)n_in; (void)out_size; (void)ws_size;

  const float* x        = (const float*)d_in[0];
  // d_in[1] = cross_input: ignored by the reference math (faithful repro).
  const float* g1       = (const float*)d_in[2];
  const float* b1       = (const float*)d_in[3];
  const float* w_qkv    = (const float*)d_in[4];
  const float* w_proj   = (const float*)d_in[5];
  const float* w_proj_b = (const float*)d_in[6];
  const float* w_scale  = (const float*)d_in[7];
  const float* c_qkv    = (const float*)d_in[8];
  const float* c_qkv_b  = (const float*)d_in[9];
  const float* c_proj   = (const float*)d_in[10];
  const float* c_proj_b = (const float*)d_in[11];
  const float* c_scale  = (const float*)d_in[12];
  const float* g2       = (const float*)d_in[13];
  const float* b2       = (const float*)d_in[14];
  const float* mlp_w1   = (const float*)d_in[15];
  const float* mlp_b1   = (const float*)d_in[16];
  const float* mlp_w2   = (const float*)d_in[17];
  const float* mlp_b2   = (const float*)d_in[18];

  const int N = 4096;  // tokens

  // Workspace partition (~44 MB total).
  char* p = (char*)d_ws;
  auto take = [&](size_t elems, size_t esz) -> void* {
    void* r = (void*)p;
    p += (elems * esz + 255) & ~(size_t)255;
    return r;
  };
  __bf16* wqkv_bf  = (__bf16*)take(256 * 768, 2);
  __bf16* wproj_bf = (__bf16*)take(256 * 256, 2);
  __bf16* cqkv_bf  = (__bf16*)take(256 * 768, 2);
  __bf16* cproj_bf = (__bf16*)take(256 * 256, 2);
  __bf16* w1_bf    = (__bf16*)take(256 * 1024, 2);
  __bf16* w2_bf    = (__bf16*)take(1024 * 256, 2);
  __bf16* winsA    = (__bf16*)take((size_t)N * 256, 2);  // LN1+shift+window, bf16
  __bf16* qkvw     = (__bf16*)take((size_t)N * 768, 2);  // window qkv
  __bf16* awA      = (__bf16*)take((size_t)N * 256, 2);  // window attn out
  float*  x1f      = (float*) take((size_t)N * 256, 4);  // shortcut + attn proj
  __bf16* x1b      = (__bf16*)take((size_t)N * 256, 2);
  __bf16* qkvc     = (__bf16*)take((size_t)N * 768, 2);  // cross qkv (+bias)
  __bf16* catt     = (__bf16*)take((size_t)N * 256, 2);  // cross attn out
  float*  x2f      = (float*) take((size_t)N * 256, 4);  // after c_proj
  float*  xn2f     = (float*) take((size_t)N * 256, 4);  // LN2 out f32
  __bf16* xn2b     = (__bf16*)take((size_t)N * 256, 2);  // LN2 out bf16
  __bf16* h_bf     = (__bf16*)take((size_t)N * 1024, 2); // relu(mlp1)

  auto cvt = [&](const float* s, __bf16* d, int n) {
    cvt_bf16<<<(n + 255) / 256, 256, 0, stream>>>(s, d, n);
  };
  cvt(w_qkv, wqkv_bf, 256 * 768);
  cvt(w_proj, wproj_bf, 256 * 256);
  cvt(c_qkv, cqkv_bf, 256 * 768);
  cvt(c_proj, cproj_bf, 256 * 256);
  cvt(mlp_w1, w1_bf, 256 * 1024);
  cvt(mlp_w2, w2_bf, 1024 * 256);

  // 1) LN1 fused with cyclic shift + window partition (row permutation).
  ln_kernel<<<N / 8, 256, 0, stream>>>(x, g1, b1, winsA, nullptr, 1);

  // 2) window qkv = winsA @ Wqkv (no bias)  [4096x256 @ 256x768]
  gemm_bf16<<<dim3(768 / 64, N / 128), 256, 0, stream>>>(
      winsA, wqkv_bf, 256, 768, nullptr, nullptr, 0, nullptr, qkvw, 0);

  // 3) window attention: 64 windows x 8 heads, seq=64.
  flash_attn<<<dim3(1, 8, 64), 128, 0, stream>>>(qkvw, awA, w_scale, 64);

  // 4) x1 = shortcut + window_reverse(unshift(aw @ Wproj + b))   (permuted rows)
  gemm_bf16<<<dim3(256 / 64, N / 128), 256, 0, stream>>>(
      awA, wproj_bf, 256, 256, w_proj_b, x, 1, x1f, x1b, 0);

  // 5) cross qkv = x1 @ Cqkv + b
  gemm_bf16<<<dim3(768 / 64, N / 128), 256, 0, stream>>>(
      x1b, cqkv_bf, 256, 768, c_qkv_b, nullptr, 0, nullptr, qkvc, 0);

  // 6) cross attention: 8 heads, seq=4096, flash (no 536MB score matrix).
  flash_attn<<<dim3(N / 64, 8, 1), 128, 0, stream>>>(qkvc, catt, c_scale, N);

  // 7) x2 = catt @ Cproj + b   (no residual per reference)
  gemm_bf16<<<dim3(256 / 64, N / 128), 256, 0, stream>>>(
      catt, cproj_bf, 256, 256, c_proj_b, nullptr, 0, x2f, nullptr, 0);

  // 8) xn2 = LN2(x2)
  ln_kernel<<<N / 8, 256, 0, stream>>>(x2f, g2, b2, xn2b, xn2f, 0);

  // 9) h = relu(xn2 @ W1 + b1)
  gemm_bf16<<<dim3(1024 / 64, N / 128), 256, 0, stream>>>(
      xn2b, w1_bf, 256, 1024, mlp_b1, nullptr, 0, nullptr, h_bf, 1);

  // 10) out = xn2 + h @ W2 + b2
  gemm_bf16<<<dim3(256 / 64, N / 128), 256, 0, stream>>>(
      h_bf, w2_bf, 1024, 256, mlp_b2, xn2f, 0, (float*)d_out, nullptr, 0);
}
